// MoELayer_25168508354997
// MI455X (gfx1250) — compile-verified, matrix operand order checked by reading, not run on hardware
//
#include <hip/hip_runtime.h>
#include <hip/hip_bf16.h>

// ---------------- problem constants ----------------
#define DH   2048     // hidden
#define FF   1408     // intermediate
#define TT   2048     // tokens = B*S
#define NE   32       // routed experts
#define ETOT 34       // routed + 2 shared
#define TOPK 6

// ---------------- GEMM tiling ----------------
#define BM   128
#define BN   64
#define BK   32
#define LDSS 40       // padded LDS row stride (ushorts) -> conflict-free b128 reads

typedef __bf16 bf16_t;
typedef bf16_t v16bf __attribute__((ext_vector_type(16)));
typedef bf16_t bf16x2 __attribute__((ext_vector_type(2)));
typedef float  v8f   __attribute__((ext_vector_type(8)));
typedef unsigned short u16x8 __attribute__((ext_vector_type(8)));

struct FragBits { u16x8 lo, hi; };

// Native RNE conversions -> v_cvt_pk_bf16_f32 (confirmed in round-3 asm).
__device__ __forceinline__ unsigned pack2bf(float lo, float hi) {
  bf16x2 v = { (bf16_t)lo, (bf16_t)hi };
  return __builtin_bit_cast(unsigned, v);
}
__device__ __forceinline__ unsigned short f2bf(float f) {
  return __builtin_bit_cast(unsigned short, (bf16_t)f);
}

// ---- gfx1250 async global->LDS copy (ASYNCcnt path) -----------------------
// Probe-confirmed signature: (v4i addrspace(1)*, v4i addrspace(3)*, imm, imm)
#if __has_builtin(__builtin_amdgcn_global_load_async_to_lds_b128)
#define ASYNC_OK 1
typedef int v4i_vs __attribute__((vector_size(16)));
typedef __attribute__((address_space(1))) v4i_vs g128_t;
typedef __attribute__((address_space(3))) v4i_vs l128_t;
__device__ __forceinline__ void async_copy16(const void* g, void* l) {
  __builtin_amdgcn_global_load_async_to_lds_b128((g128_t*)g, (l128_t*)l, 0, 0);
}
__device__ __forceinline__ void async_wait0() {
#if __has_builtin(__builtin_amdgcn_s_wait_asynccnt)
  __builtin_amdgcn_s_wait_asynccnt(0);
#else
  asm volatile("s_wait_asynccnt 0x0" ::: "memory");
#endif
}
#endif

// Load one 16x32 bf16 A/B fragment (ISA 16-bit striped layout) from an LDS
// tile stored row-major with LDSS stride.
__device__ __forceinline__ v16bf ldsFrag(const unsigned short* tile, int rowBase, int lane) {
  const int r = lane & 15, half = lane >> 4;
  const unsigned short* p = tile + (rowBase + r) * LDSS + half * 8;
  FragBits fb;
  fb.lo = *(const u16x8*)(p);        // ds_load_b128
  fb.hi = *(const u16x8*)(p + 16);   // ds_load_b128
  return __builtin_bit_cast(v16bf, fb);
}

// ---------------- kernel 1: router (one wave32 per token, lane = expert) ----
__global__ __launch_bounds__(128) void moe_router(
    const float* __restrict__ x, const float* __restrict__ wr,
    int* __restrict__ counts, int* __restrict__ token_ids, float* __restrict__ tok_w)
{
  const int t    = blockIdx.x * 4 + (threadIdx.x >> 5);
  const int lane = threadIdx.x & 31;

  float acc = 0.f;
  const float* xr = x + (size_t)t * DH;
  for (int d0 = 0; d0 < DH; d0 += 32) {
    float xv = xr[d0 + lane];
    #pragma unroll 8
    for (int j = 0; j < 32; ++j) {
      float xs = __shfl(xv, j, 32);
      acc = fmaf(xs, wr[(d0 + j) * NE + lane], acc);
    }
  }
  float m = acc;
  for (int off = 16; off > 0; off >>= 1) m = fmaxf(m, __shfl_xor(m, off, 32));
  float p = __expf(acc - m);
  float s = p;
  for (int off = 16; off > 0; off >>= 1) s += __shfl_xor(s, off, 32);
  p /= s;

  float q = p, tsum = 0.f;
  int sel = 0;
  #pragma unroll
  for (int i = 0; i < TOPK; ++i) {
    float v = q; int idx = lane;
    for (int off = 16; off > 0; off >>= 1) {
      float ov = __shfl_xor(v, off, 32);
      int   oi = __shfl_xor(idx, off, 32);
      if (ov > v || (ov == v && oi < idx)) { v = ov; idx = oi; }
    }
    tsum += v;
    if (lane == idx) { sel = 1; q = -1e30f; }
  }
  if (sel) {
    int pos = atomicAdd(&counts[lane], 1);
    token_ids[lane * TT + pos] = t;
    tok_w[lane * TT + pos]     = p / tsum;
  }
}

// ---------------- kernel 2: row offsets for the h workspace ----------------
__global__ void moe_offsets(const int* __restrict__ counts, int* __restrict__ row_off)
{
  if (threadIdx.x == 0) {
    int off = 0;
    for (int e = 0; e < ETOT; ++e) { row_off[e] = off; off += (e < NE) ? counts[e] : TT; }
    row_off[ETOT] = off;
  }
}

// ---------------- kernel 3: gate+up GEMM, fused SiLU, bf16 h out -----------
// Double-buffered: stage(k) -> barrier -> fetch(k+1 into regs) -> compute(k).
__global__ __launch_bounds__(256) void moe_gemm_gateup(
    const float* __restrict__ x,
    const float* __restrict__ wg, const float* __restrict__ wu,
    const float* __restrict__ sg, const float* __restrict__ su,
    const int* __restrict__ counts, const int* __restrict__ row_off,
    const int* __restrict__ token_ids,
    unsigned short* __restrict__ h)
{
  const int e  = blockIdx.z;
  const bool routed = (e < NE);
  const int Te = routed ? counts[e] : TT;
  const int m0 = blockIdx.x * BM;
  if (m0 >= Te) return;
  const int n0 = blockIdx.y * BN;

  const float* Wg = routed ? wg + (size_t)e * DH * FF : sg + (size_t)(e - NE) * DH * FF;
  const float* Wu = routed ? wu + (size_t)e * DH * FF : su + (size_t)(e - NE) * DH * FF;
  const int* list = token_ids + (size_t)(routed ? e : 0) * TT;

  __shared__ unsigned short As [2][BM * LDSS];
  __shared__ unsigned short Bgs[2][BN * LDSS];
  __shared__ unsigned short Bus[2][BN * LDSS];

  const int tid  = threadIdx.x;
  const int lane = tid & 31;
  const int wid  = tid >> 5;
  const int wm   = (wid & 3) * 32;
  const int wn   = (wid >> 2) * 32;
  const int bk2  = (tid >> 4) * 2;
  const int bn4  = (tid & 15) * 4;

  // K-invariant per-thread A row pointers (clamped; garbage rows never stored)
  const float* aptr[4];
  #pragma unroll
  for (int i = 0; i < 4; ++i) {
    int lin = tid * 4 + i * 1024;
    int gm  = m0 + (lin >> 5);
    int gmc = gm < Te ? gm : Te - 1;
    int tok = routed ? list[gmc] : gmc;
    aptr[i] = x + (size_t)tok * DH + (lin & 31);
  }
  const float* bgp = Wg + (size_t)bk2 * FF + n0 + bn4;
  const float* bup = Wu + (size_t)bk2 * FF + n0 + bn4;

  float4 ra[4], rg0, rg1, ru0, ru1;
  auto fetchT = [&](int k0) {
    #pragma unroll
    for (int i = 0; i < 4; ++i) ra[i] = *(const float4*)(aptr[i] + k0);
    const float* bg = bgp + (size_t)k0 * FF;
    const float* bu = bup + (size_t)k0 * FF;
    rg0 = *(const float4*)bg; rg1 = *(const float4*)(bg + FF);
    ru0 = *(const float4*)bu; ru1 = *(const float4*)(bu + FF);
  };
  auto stageT = [&](int buf) {
    #pragma unroll
    for (int i = 0; i < 4; ++i) {
      int lin = tid * 4 + i * 1024;
      int r = lin >> 5, c = lin & 31;
      unsigned long long pk = (unsigned long long)pack2bf(ra[i].x, ra[i].y)
                            | ((unsigned long long)pack2bf(ra[i].z, ra[i].w) << 32);
      *(unsigned long long*)(&As[buf][r * LDSS + c]) = pk;   // ds_store_b64
    }
    *(unsigned*)(&Bgs[buf][(bn4+0)*LDSS + bk2]) = pack2bf(rg0.x, rg1.x);
    *(unsigned*)(&Bgs[buf][(bn4+1)*LDSS + bk2]) = pack2bf(rg0.y, rg1.y);
    *(unsigned*)(&Bgs[buf][(bn4+2)*LDSS + bk2]) = pack2bf(rg0.z, rg1.z);
    *(unsigned*)(&Bgs[buf][(bn4+3)*LDSS + bk2]) = pack2bf(rg0.w, rg1.w);
    *(unsigned*)(&Bus[buf][(bn4+0)*LDSS + bk2]) = pack2bf(ru0.x, ru1.x);
    *(unsigned*)(&Bus[buf][(bn4+1)*LDSS + bk2]) = pack2bf(ru0.y, ru1.y);
    *(unsigned*)(&Bus[buf][(bn4+2)*LDSS + bk2]) = pack2bf(ru0.z, ru1.z);
    *(unsigned*)(&Bus[buf][(bn4+3)*LDSS + bk2]) = pack2bf(ru0.w, ru1.w);
  };

  const v8f zero = {0.f,0.f,0.f,0.f,0.f,0.f,0.f,0.f};
  v8f cg[2][2] = {{zero, zero}, {zero, zero}};
  v8f cu[2][2] = {{zero, zero}, {zero, zero}};

  const int NK = DH / BK;
  fetchT(0);
  for (int k = 0; k < NK; ++k) {
    const int buf = k & 1;
    stageT(buf);
    __syncthreads();
    if (k + 1 < NK) fetchT((k + 1) * BK);   // global loads in flight over WMMAs

    v16bf a0  = ldsFrag(As[buf],  wm,      lane);
    v16bf a1  = ldsFrag(As[buf],  wm + 16, lane);
    v16bf bg0 = ldsFrag(Bgs[buf], wn,      lane);
    v16bf bg1 = ldsFrag(Bgs[buf], wn + 16, lane);
    v16bf bu0 = ldsFrag(Bus[buf], wn,      lane);
    v16bf bu1 = ldsFrag(Bus[buf], wn + 16, lane);

    cg[0][0] = __builtin_amdgcn_wmma_f32_16x16x32_bf16(false, a0, false, bg0, (short)0, cg[0][0], false, false);
    cg[0][1] = __builtin_amdgcn_wmma_f32_16x16x32_bf16(false, a0, false, bg1, (short)0, cg[0][1], false, false);
    cg[1][0] = __builtin_amdgcn_wmma_f32_16x16x32_bf16(false, a1, false, bg0, (short)0, cg[1][0], false, false);
    cg[1][1] = __builtin_amdgcn_wmma_f32_16x16x32_bf16(false, a1, false, bg1, (short)0, cg[1][1], false, false);
    cu[0][0] = __builtin_amdgcn_wmma_f32_16x16x32_bf16(false, a0, false, bu0, (short)0, cu[0][0], false, false);
    cu[0][1] = __builtin_amdgcn_wmma_f32_16x16x32_bf16(false, a0, false, bu1, (short)0, cu[0][1], false, false);
    cu[1][0] = __builtin_amdgcn_wmma_f32_16x16x32_bf16(false, a1, false, bu0, (short)0, cu[1][0], false, false);
    cu[1][1] = __builtin_amdgcn_wmma_f32_16x16x32_bf16(false, a1, false, bu1, (short)0, cu[1][1], false, false);
  }

  // epilogue: h = silu(g) * u, store bf16
  const int rowoff = row_off[e];
  #pragma unroll
  for (int mi = 0; mi < 2; ++mi) {
    #pragma unroll
    for (int ni = 0; ni < 2; ++ni) {
      #pragma unroll
      for (int v = 0; v < 8; ++v) {
        int r  = wm + mi * 16 + (lane >> 4) * 8 + v;
        int gm = m0 + r;
        if (gm < Te) {
          int col = n0 + wn + ni * 16 + (lane & 15);
          float g = cg[mi][ni][v];
          float u = cu[mi][ni][v];
          float hv = (g / (1.f + __expf(-g))) * u;
          h[(size_t)(rowoff + gm) * FF + col] = f2bf(hv);
        }
      }
    }
  }
}

// ---------------- kernel 4: down GEMM + weighted scatter-accumulate --------
// A tile is bf16 already -> async global->LDS engine, double buffered.
__global__ __launch_bounds__(256) void moe_gemm_down(
    const float* __restrict__ wd, const float* __restrict__ sd,
    const int* __restrict__ counts, const int* __restrict__ row_off,
    const int* __restrict__ token_ids, const float* __restrict__ tok_w,
    const unsigned short* __restrict__ h,
    float* __restrict__ out)
{
  const int e  = blockIdx.z;
  const bool routed = (e < NE);
  const int Te = routed ? counts[e] : TT;
  const int m0 = blockIdx.x * BM;
  if (m0 >= Te) return;
  const int n0 = blockIdx.y * BN;

  const float* Wd   = routed ? wd + (size_t)e * FF * DH : sd + (size_t)(e - NE) * FF * DH;
  const int*   list = token_ids + (size_t)(routed ? e : 0) * TT;
  const float* wrow = tok_w    + (size_t)(routed ? e : 0) * TT;
  const int rowoff  = row_off[e];

  __shared__ unsigned short As[2][BM * LDSS];
  __shared__ unsigned short Bs[2][BN * LDSS];

  const int tid  = threadIdx.x;
  const int lane = tid & 31;
  const int wid  = tid >> 5;
  const int wm   = (wid & 3) * 32;
  const int wn   = (wid >> 2) * 32;
  const int bk2  = (tid >> 4) * 2;
  const int bn4  = (tid & 15) * 4;

  // K-invariant per-thread A source pointers (clamped rows -> always valid)
  const unsigned short* hptr[2];
  int lc[2], lr[2];
  #pragma unroll
  for (int i = 0; i < 2; ++i) {
    int lin = tid * 8 + i * 2048;
    lr[i] = lin >> 5; lc[i] = lin & 31;
    int gm  = m0 + lr[i];
    int gmc = gm < Te ? gm : Te - 1;
    hptr[i] = h + (size_t)(rowoff + gmc) * FF + lc[i];
  }
  auto issueA = [&](int k0, int buf) {
    #pragma unroll
    for (int i = 0; i < 2; ++i) {
      unsigned short* dst = &As[buf][lr[i] * LDSS + lc[i]];
#ifdef ASYNC_OK
      async_copy16(hptr[i] + k0, dst);        // global_load_async_to_lds_b128
#else
      *(u16x8*)dst = *(const u16x8*)(hptr[i] + k0);
#endif
    }
  };

  const float* bp = Wd + (size_t)bk2 * DH + n0 + bn4;
  float4 rb0, rb1;
  auto fetchB = [&](int k0) {
    const float* b = bp + (size_t)k0 * DH;
    rb0 = *(const float4*)b;
    rb1 = *(const float4*)(b + DH);
  };
  auto stageB = [&](int buf) {
    *(unsigned*)(&Bs[buf][(bn4+0)*LDSS + bk2]) = pack2bf(rb0.x, rb1.x);
    *(unsigned*)(&Bs[buf][(bn4+1)*LDSS + bk2]) = pack2bf(rb0.y, rb1.y);
    *(unsigned*)(&Bs[buf][(bn4+2)*LDSS + bk2]) = pack2bf(rb0.z, rb1.z);
    *(unsigned*)(&Bs[buf][(bn4+3)*LDSS + bk2]) = pack2bf(rb0.w, rb1.w);
  };

  const v8f zero = {0.f,0.f,0.f,0.f,0.f,0.f,0.f,0.f};
  v8f acc[2][2] = {{zero, zero}, {zero, zero}};

  const int NK = FF / BK;            // 44
  issueA(0, 0);
  fetchB(0);
  for (int k = 0; k < NK; ++k) {
    const int buf = k & 1;
    stageB(buf);
#ifdef ASYNC_OK
    async_wait0();                   // our tile-k async copies have landed
#endif
    __syncthreads();
    if (k + 1 < NK) {                // next tile in flight over the WMMAs
      issueA((k + 1) * BK, buf ^ 1);
      fetchB((k + 1) * BK);
    }

    v16bf a0 = ldsFrag(As[buf], wm,      lane);
    v16bf a1 = ldsFrag(As[buf], wm + 16, lane);
    v16bf b0 = ldsFrag(Bs[buf], wn,      lane);
    v16bf b1 = ldsFrag(Bs[buf], wn + 16, lane);

    acc[0][0] = __builtin_amdgcn_wmma_f32_16x16x32_bf16(false, a0, false, b0, (short)0, acc[0][0], false, false);
    acc[0][1] = __builtin_amdgcn_wmma_f32_16x16x32_bf16(false, a0, false, b1, (short)0, acc[0][1], false, false);
    acc[1][0] = __builtin_amdgcn_wmma_f32_16x16x32_bf16(false, a1, false, b0, (short)0, acc[1][0], false, false);
    acc[1][1] = __builtin_amdgcn_wmma_f32_16x16x32_bf16(false, a1, false, b1, (short)0, acc[1][1], false, false);
  }

  // epilogue: out[tok] += w_e * y   (fp32 global atomic add)
  #pragma unroll
  for (int mi = 0; mi < 2; ++mi) {
    #pragma unroll
    for (int ni = 0; ni < 2; ++ni) {
      #pragma unroll
      for (int v = 0; v < 8; ++v) {
        int r  = wm + mi * 16 + (lane >> 4) * 8 + v;
        int gm = m0 + r;
        if (gm < Te) {
          int tok; float w;
          if (routed) { tok = list[gm]; w = wrow[gm]; }
          else        { tok = gm;       w = 1.f; }
          int col = n0 + wn + ni * 16 + (lane & 15);
          __hip_atomic_fetch_add(out + (size_t)tok * DH + col, acc[mi][ni][v] * w,
                                 __ATOMIC_RELAXED, __HIP_MEMORY_SCOPE_AGENT);
        }
      }
    }
  }
}

// ---------------- workspace layout ----------------
#define WS_COUNTS 0
#define WS_ROWOFF 256
#define WS_TOKIDS 512
#define WS_TOKW   (512 + NE * TT * 4)
#define WS_H      525312

extern "C" void kernel_launch(void* const* d_in, const int* in_sizes, int n_in,
                              void* d_out, int out_size, void* d_ws, size_t ws_size,
                              hipStream_t stream) {
  const float* x  = (const float*)d_in[0];
  const float* wr = (const float*)d_in[1];
  const float* wg = (const float*)d_in[2];
  const float* wu = (const float*)d_in[3];
  const float* wd = (const float*)d_in[4];
  const float* sg = (const float*)d_in[5];
  const float* su = (const float*)d_in[6];
  const float* sd = (const float*)d_in[7];
  float* out = (float*)d_out;

  char* ws = (char*)d_ws;
  int*   counts    = (int*)  (ws + WS_COUNTS);
  int*   row_off   = (int*)  (ws + WS_ROWOFF);
  int*   token_ids = (int*)  (ws + WS_TOKIDS);
  float* tok_w     = (float*)(ws + WS_TOKW);
  unsigned short* h = (unsigned short*)(ws + WS_H);

  (void)hipMemsetAsync(counts, 0, 128, stream);
  (void)hipMemsetAsync(out, 0, (size_t)out_size * sizeof(float), stream);

  moe_router<<<TT / 4, 128, 0, stream>>>(x, wr, counts, token_ids, tok_w);
  moe_offsets<<<1, 32, 0, stream>>>(counts, row_off);

  dim3 g1(TT / BM, FF / BN, ETOT);   // (16, 22, 34)
  moe_gemm_gateup<<<g1, 256, 0, stream>>>(x, wg, wu, sg, su, counts, row_off, token_ids, h);

  dim3 g2(TT / BM, DH / BN, ETOT);   // (16, 32, 34)
  moe_gemm_down<<<g2, 256, 0, stream>>>(wd, sd, counts, row_off, token_ids, tok_w, h, out);
}
